// RCNNHead_29437705847002
// MI455X (gfx1250) — matrix operand order repaired and checked by reading, not compiled, not run on hardware
//
#include <hip/hip_runtime.h>
#include <hip/hip_bf16.h>
#include <math.h>

// Problem constants (match reference)
#define BB   8
#define NNp  512
#define CC   256
#define HW   49          // 7x7
#define MM   (BB*NNp)    // 4096
#define CQ   64

typedef __attribute__((ext_vector_type(16))) __bf16 v16bf;
typedef __attribute__((ext_vector_type(8)))  float  v8f;

union BFU { v16bf v; unsigned int u[8]; };

__device__ __forceinline__ unsigned short f2bf(float f) {
  unsigned int x = __float_as_uint(f);
  x += 0x7FFFu + ((x >> 16) & 1u);           // round-to-nearest-even
  return (unsigned short)(x >> 16);
}
__device__ __forceinline__ unsigned int pack2bf(float lo, float hi) {
  return ((unsigned int)f2bf(hi) << 16) | (unsigned int)f2bf(lo);
}
__device__ __forceinline__ float bf2f(unsigned short u) {
  return __uint_as_float(((unsigned int)u) << 16);
}

// ---------------------------------------------------------------------------
// Prep: bf16 weight conversion.  Wcomb = [Wcor (256x256) ; W1[:, :256] (64x256)]
// ---------------------------------------------------------------------------
__global__ void k_prep_weights(const float* __restrict__ Wcor,
                               const float* __restrict__ W1,
                               const float* __restrict__ fc1w,
                               const float* __restrict__ fc2w,
                               unsigned short* __restrict__ wcomb,
                               unsigned short* __restrict__ fc1b16,
                               unsigned short* __restrict__ fc2b16) {
  int idx = blockIdx.x * 256 + threadIdx.x;
  if (idx < 320 * 256) {
    int row = idx >> 8, c = idx & 255;
    float v = (row < 256) ? Wcor[row * 256 + c] : W1[(row - 256) * 512 + c];
    wcomb[idx] = f2bf(v);
  }
  if (idx < 256 * 512) fc1b16[idx] = f2bf(fc1w[idx]);
  if (idx < 256 * 256) fc2b16[idx] = f2bf(fc2w[idx]);
}

// s_mean[b][c] = mean over 49 of x_s
__global__ void k_prep_smean(const float* __restrict__ xs, float* __restrict__ smean) {
  int b = blockIdx.x, c = threadIdx.x;
  const float* p = xs + (size_t)b * CC * HW + c * HW;
  float s = 0.f;
  for (int h = 0; h < HW; ++h) s += p[h];
  smean[b * 256 + c] = s * (1.f / 49.f);
}

// y_s[b][o][hw] = (Wcor @ x_s[b])[o][hw], padded to 64 wide with zeros
__global__ void k_prep_ys(const float* __restrict__ Wcor,
                          const float* __restrict__ xs, float* __restrict__ ys) {
  int bo = blockIdx.x;               // b*256 + o
  int b = bo >> 8, o = bo & 255;
  int hw = threadIdx.x;              // 0..63
  float s = 0.f;
  if (hw < HW) {
    const float* wr = Wcor + o * 256;
    const float* xp = xs + (size_t)b * CC * HW;
    for (int c = 0; c < 256; ++c) s += wr[c] * xp[c * HW + hw];
  }
  ys[(size_t)bo * 64 + hw] = s;
}

// w1s[b][o][hw] = sum_c W1[o][256+c] * x_s[b][c][hw]
__global__ void k_prep_w1s(const float* __restrict__ W1,
                           const float* __restrict__ xs, float* __restrict__ w1s) {
  int bo = blockIdx.x;               // b*64 + o
  int b = bo >> 6, o = bo & 63;
  int hw = threadIdx.x;
  if (hw >= HW) return;
  const float* wr = W1 + o * 512 + 256;
  const float* xp = xs + (size_t)b * CC * HW;
  float s = 0.f;
  for (int c = 0; c < 256; ++c) s += wr[c] * xp[c * HW + hw];
  w1s[bo * HW + hw] = s;
}

// ---------------------------------------------------------------------------
// Main per-proposal kernel: fused [Wcor;W1q] GEMM via WMMA + correlation
// reduction + 1x1/pool/conv3x3/pool path + score_pr/deltas.
// One block per proposal m; 256 threads = 8 waves; 40 dual-ntile WMMA jobs
// (A fragment shared between the two n-tiles of each job).
// ---------------------------------------------------------------------------
__global__ __launch_bounds__(256) void k_main(
    const float* __restrict__ bfq, const unsigned short* __restrict__ wcomb,
    const float* __restrict__ ys, const float* __restrict__ w1s,
    const float* __restrict__ W2, const float* __restrict__ W3,
    const float* __restrict__ scorw, const float* __restrict__ scorb,
    const float* __restrict__ sprw, const float* __restrict__ sprb,
    const float* __restrict__ dprw, const float* __restrict__ dprb,
    float* __restrict__ qmean_g, float* __restrict__ spart_g,
    float* __restrict__ deltas_g) {
  __shared__ unsigned short xqT[64 * 258];   // [n=hw pad 64][k=c], pitch 258 (129 dw)
  __shared__ float convbuf[64 * 64];         // W1q @ x_q output (rows 256..319)
  __shared__ float corsum[256];
  __shared__ float p1s[64 * 25];
  __shared__ float c2s[64 * 9];
  __shared__ float red[256];

  const int t = threadIdx.x;
  const int m = blockIdx.x;
  const int b = m >> 9;

  corsum[t] = 0.f;
  unsigned int* xz = (unsigned int*)xqT;
  for (int i = t; i < 64 * 129; i += 256) xz[i] = 0u;
  __syncthreads();

  // Stage x_q[m] -> LDS (transposed, bf16), coalesced global reads
  const float* gq = bfq + (size_t)m * (CC * HW);
  #pragma unroll 1
  for (int i = 0; i < 49; ++i) {
    int idx = i * 256 + t;                  // 49*256 == 256*49 elements exactly
    int c = idx / HW, hw = idx % HW;
    float v = gq[idx];
    __builtin_prefetch(gq + idx + 2048, 0, 0);
    xqT[hw * 258 + c] = f2bf(v);
  }
  __syncthreads();

  // q_mean for channel t (sum the staged bf16 tile; matches bf16 fc1 input)
  {
    float s = 0.f;
    #pragma unroll 7
    for (int hw = 0; hw < HW; ++hw) s += bf2f(xqT[hw * 258 + t]);
    qmean_g[(size_t)m * 256 + t] = s * (1.f / 49.f);
  }

  const int lane = t & 31, wid = t >> 5;
  const int lm = lane & 15, half = lane >> 4;
  const unsigned int* wcu = (const unsigned int*)wcomb;
  const unsigned int* lqu = (const unsigned int*)xqT;

  // GEMM: Wcomb(320x256) x xqT(256x64) ; 20 mtiles x 2 ntile-pairs = 40 jobs
  for (int dj = wid; dj < 40; dj += 8) {
    int mt = dj >> 1, np = dj & 1;
    v8f acc0, acc1;
    #pragma unroll
    for (int r = 0; r < 8; ++r) { acc0[r] = 0.f; acc1[r] = 0.f; }
    int arow = mt * 16 + lm;
    int n0 = np * 32 + lm;                  // first n-tile row for B
    #pragma unroll 2
    for (int ks = 0; ks < 8; ++ks) {
      BFU aU, b0U, b1U;
      int kba = (half ? 8 : 0) + ks * 32;
      #pragma unroll
      for (int j = 0; j < 8; ++j) {
        int k = kba + ((j >> 2) << 4) + ((j & 3) << 1);
        aU.u[j] = wcu[arow * 128 + (k >> 1)];
      }
      int kbb = (half ? 8 : 0) + ks * 32;
      #pragma unroll
      for (int j = 0; j < 8; ++j) {
        int k = kbb + ((j >> 2) << 4) + ((j & 3) << 1);
        b0U.u[j] = lqu[n0 * 129 + (k >> 1)];
        b1U.u[j] = lqu[(n0 + 16) * 129 + (k >> 1)];
      }
      acc0 = __builtin_amdgcn_wmma_f32_16x16x32_bf16(
          false, aU.v, false, b0U.v, (short)0, acc0, false, false);
      acc1 = __builtin_amdgcn_wmma_f32_16x16x32_bf16(
          false, aU.v, false, b1U.v, (short)0, acc1, false, false);
    }
    if (mt < 16) {
      // correlation rows: dot with y_s; combine both n-tiles, one reduce+atomic
      #pragma unroll
      for (int r = 0; r < 8; ++r) {
        int o   = mt * 16 + r + 8 * half;
        int hw0 = np * 32 + lm;
        const float* yp = ys + ((size_t)(b * 256 + o)) * 64;
        float part = acc0[r] * yp[hw0] + acc1[r] * yp[hw0 + 16];
        part += __shfl_xor(part, 1, 32);
        part += __shfl_xor(part, 2, 32);
        part += __shfl_xor(part, 4, 32);
        part += __shfl_xor(part, 8, 32);
        if (lm == 0) atomicAdd(&corsum[o], part);
      }
    } else {
      #pragma unroll
      for (int r = 0; r < 8; ++r) {
        int o = (mt - 16) * 16 + r + 8 * half;
        int c0 = np * 32 + lm;
        convbuf[o * 64 + c0]      = acc0[r];
        convbuf[o * 64 + c0 + 16] = acc1[r];
      }
    }
  }
  __syncthreads();

  // score_cor = scor_w . relu(corsum) + scor_b
  red[t] = fmaxf(corsum[t], 0.f) * scorw[t];
  __syncthreads();
  for (int s = 128; s > 0; s >>= 1) { if (t < s) red[t] += red[t + s]; __syncthreads(); }
  float score_cor = red[0] + scorb[0];
  __syncthreads();

  // x1 = relu(convbuf + w1s) ; avgpool3 7->5
  for (int idx = t; idx < 64 * 25; idx += 256) {
    int o = idx / 25, pos = idx % 25, py = pos / 5, px = pos % 5;
    float s = 0.f;
    #pragma unroll
    for (int dy = 0; dy < 3; ++dy)
      #pragma unroll
      for (int dx = 0; dx < 3; ++dx) {
        int hw = (py + dy) * 7 + (px + dx);
        s += fmaxf(convbuf[o * 64 + hw] + w1s[(b * 64 + o) * HW + hw], 0.f);
      }
    p1s[idx] = s * (1.f / 9.f);
  }
  __syncthreads();

  // conv3x3 (64->64), valid 5->3, relu
  for (int idx = t; idx < 64 * 9; idx += 256) {
    int o = idx / 9, pos = idx % 9, cy = pos / 3, cx = pos % 3;
    float s = 0.f;
    for (int i = 0; i < 64; ++i) {
      #pragma unroll
      for (int dy = 0; dy < 3; ++dy)
        #pragma unroll
        for (int dx = 0; dx < 3; ++dx)
          s += W2[((o * 64 + i) * 3 + dy) * 3 + dx] *
               p1s[i * 25 + (cy + dy) * 5 + (cx + dx)];
    }
    c2s[idx] = fmaxf(s, 0.f);
  }
  __syncthreads();

  // W3 (256x64) + relu, then avgpool 3x3->1 ; thread t owns output channel t
  float xs = 0.f;
  for (int pos = 0; pos < 9; ++pos) {
    float a3 = 0.f;
    for (int i = 0; i < 64; ++i) a3 += W3[t * 64 + i] * c2s[i * 9 + pos];
    xs += fmaxf(a3, 0.f);
  }
  float xv = xs * (1.f / 9.f);

  red[t] = xv * sprw[t];
  __syncthreads();
  for (int s = 128; s > 0; s >>= 1) { if (t < s) red[t] += red[t + s]; __syncthreads(); }
  if (t == 0) spart_g[m] = red[0] + sprb[0] + score_cor;
  __syncthreads();

  for (int kd = 0; kd < 4; ++kd) {
    red[t] = xv * dprw[kd * 256 + t];
    __syncthreads();
    for (int s = 128; s > 0; s >>= 1) { if (t < s) red[t] += red[t + s]; __syncthreads(); }
    if (t == 0) deltas_g[m * 4 + kd] = red[0] + dprb[kd];
    __syncthreads();
  }
}

// ---------------------------------------------------------------------------
// fc1: h1 = relu([q_mean, s_mean] @ fc1_w^T + b)   (4096x256x512 WMMA GEMM)
// Dual n-tile per wave; A fragment (built from f32 means) shared.
// ---------------------------------------------------------------------------
__global__ __launch_bounds__(256) void k_fc1(
    const float* __restrict__ qmean, const float* __restrict__ smean,
    const unsigned short* __restrict__ fc1b16, const float* __restrict__ fc1bias,
    unsigned short* __restrict__ h1b16) {
  int t = threadIdx.x, lane = t & 31, wid = t >> 5;
  int lm = lane & 15, half = lane >> 4;
  int dt = blockIdx.x * 8 + wid;          // 0..2047 : 256 mtiles x 8 ntile-pairs
  int mt = dt >> 3, np = dt & 7;
  int arow = mt * 16 + lm;
  int bb = arow >> 9;
  int n0 = np * 32 + lm;
  const unsigned int* fu = (const unsigned int*)fc1b16;
  v8f acc0, acc1;
  #pragma unroll
  for (int r = 0; r < 8; ++r) { acc0[r] = 0.f; acc1[r] = 0.f; }
  #pragma unroll 2
  for (int ks = 0; ks < 16; ++ks) {
    BFU aU, b0U, b1U;
    int kba = (half ? 8 : 0) + ks * 32;
    // k<256 vs k>=256 is uniform per k-step: hoist source pointer
    const float* src = (kba < 256) ? (qmean + (size_t)arow * 256)
                                   : (smean + bb * 256 - 256);
    #pragma unroll
    for (int j = 0; j < 8; ++j) {
      int k = kba + ((j >> 2) << 4) + ((j & 3) << 1);
      aU.u[j] = pack2bf(src[k], src[k + 1]);
    }
    int kbb = (half ? 8 : 0) + ks * 32;
    #pragma unroll
    for (int j = 0; j < 8; ++j) {
      int k = kbb + ((j >> 2) << 4) + ((j & 3) << 1);
      b0U.u[j] = fu[n0 * 256 + (k >> 1)];
      b1U.u[j] = fu[(n0 + 16) * 256 + (k >> 1)];
    }
    acc0 = __builtin_amdgcn_wmma_f32_16x16x32_bf16(
        false, aU.v, false, b0U.v, (short)0, acc0, false, false);
    acc1 = __builtin_amdgcn_wmma_f32_16x16x32_bf16(
        false, aU.v, false, b1U.v, (short)0, acc1, false, false);
  }
  float bias0 = fc1bias[n0], bias1 = fc1bias[n0 + 16];
  #pragma unroll
  for (int r = 0; r < 8; ++r) {
    int mrow = mt * 16 + r + 8 * half;
    h1b16[(size_t)mrow * 256 + n0]      = f2bf(fmaxf(acc0[r] + bias0, 0.f));
    h1b16[(size_t)mrow * 256 + n0 + 16] = f2bf(fmaxf(acc1[r] + bias1, 0.f));
  }
}

// ---------------------------------------------------------------------------
// fc2 + score_fc epilogue: h2=relu(h1@fc2^T+b); score_fc += sfc_w . h2
// ---------------------------------------------------------------------------
__global__ __launch_bounds__(256) void k_fc2(
    const unsigned short* __restrict__ h1b16, const unsigned short* __restrict__ fc2b16,
    const float* __restrict__ fc2bias, const float* __restrict__ sfcw,
    float* __restrict__ scorefc) {
  int t = threadIdx.x, lane = t & 31, wid = t >> 5;
  int lm = lane & 15, half = lane >> 4;
  int dt = blockIdx.x * 8 + wid;          // 0..2047
  int mt = dt >> 3, np = dt & 7;
  int arow = mt * 16 + lm;
  int n0 = np * 32 + lm;
  const unsigned int* au = (const unsigned int*)h1b16;
  const unsigned int* bu = (const unsigned int*)fc2b16;
  v8f acc0, acc1;
  #pragma unroll
  for (int r = 0; r < 8; ++r) { acc0[r] = 0.f; acc1[r] = 0.f; }
  #pragma unroll 2
  for (int ks = 0; ks < 8; ++ks) {
    BFU aU, b0U, b1U;
    int kba = (half ? 8 : 0) + ks * 32;
    #pragma unroll
    for (int j = 0; j < 8; ++j) {
      int k = kba + ((j >> 2) << 4) + ((j & 3) << 1);
      aU.u[j] = au[arow * 128 + (k >> 1)];
    }
    int kbb = (half ? 8 : 0) + ks * 32;
    #pragma unroll
    for (int j = 0; j < 8; ++j) {
      int k = kbb + ((j >> 2) << 4) + ((j & 3) << 1);
      b0U.u[j] = bu[n0 * 128 + (k >> 1)];
      b1U.u[j] = bu[(n0 + 16) * 128 + (k >> 1)];
    }
    acc0 = __builtin_amdgcn_wmma_f32_16x16x32_bf16(
        false, aU.v, false, b0U.v, (short)0, acc0, false, false);
    acc1 = __builtin_amdgcn_wmma_f32_16x16x32_bf16(
        false, aU.v, false, b1U.v, (short)0, acc1, false, false);
  }
  float bias0 = fc2bias[n0], w0 = sfcw[n0];
  float bias1 = fc2bias[n0 + 16], w1 = sfcw[n0 + 16];
  #pragma unroll
  for (int r = 0; r < 8; ++r) {
    float v = fmaxf(acc0[r] + bias0, 0.f) * w0 +
              fmaxf(acc1[r] + bias1, 0.f) * w1;
    v += __shfl_xor(v, 1, 32);
    v += __shfl_xor(v, 2, 32);
    v += __shfl_xor(v, 4, 32);
    v += __shfl_xor(v, 8, 32);
    if (lm == 0) atomicAdd(&scorefc[mt * 16 + r + 8 * half], v);
  }
}

__global__ void k_init_scorefc(const float* __restrict__ sfcb, float* __restrict__ scorefc) {
  int i = blockIdx.x * 256 + threadIdx.x;
  if (i < MM) scorefc[i] = sfcb[0];
}

// ---------------------------------------------------------------------------
// Decode boxes + sigmoid scores
// ---------------------------------------------------------------------------
__global__ void k_decode(const float* __restrict__ spart, const float* __restrict__ scorefc,
                         const float* __restrict__ deltas, const float* __restrict__ proposals,
                         const int* __restrict__ imgsz, float* __restrict__ out) {
  int m = blockIdx.x * 256 + threadIdx.x;
  if (m >= MM) return;
  float sc = spart[m] + scorefc[m];
  float prob = 1.f / (1.f + expf(-sc));
  float x1 = proposals[m * 4 + 0], y1 = proposals[m * 4 + 1];
  float x2 = proposals[m * 4 + 2], y2 = proposals[m * 4 + 3];
  float w = x2 - x1, h = y2 - y1;
  float cx = x1 + 0.5f * w, cy = y1 + 0.5f * h;
  float dx = deltas[m * 4 + 0], dy = deltas[m * 4 + 1];
  const float CLIP = 4.135166556742356f;
  float dw = fminf(deltas[m * 4 + 2], CLIP);
  float dh = fminf(deltas[m * 4 + 3], CLIP);
  float pcx = dx * w + cx, pcy = dy * h + cy;
  float pw = expf(dw) * w, ph = expf(dh) * h;
  float Wc = (float)imgsz[1], Hc = (float)imgsz[0];
  out[m * 4 + 0] = fminf(fmaxf(pcx - 0.5f * pw, 0.f), Wc);
  out[m * 4 + 1] = fminf(fmaxf(pcy - 0.5f * ph, 0.f), Hc);
  out[m * 4 + 2] = fminf(fmaxf(pcx + 0.5f * pw, 0.f), Wc);
  out[m * 4 + 3] = fminf(fmaxf(pcy + 0.5f * ph, 0.f), Hc);
  out[MM * 4 + m] = prob;
}

// ---------------------------------------------------------------------------
// Per-batch NMS (stable argsort via ranking + sequential suppression)
// ---------------------------------------------------------------------------
__global__ __launch_bounds__(512) void k_nms(float* __restrict__ out) {
  __shared__ float sc[512];
  __shared__ float bx[512 * 4];
  __shared__ int   ordIdx[512];
  __shared__ float sb[512 * 4];
  __shared__ float sArea[512];
  __shared__ int   keepS[512];
  int i = threadIdx.x;
  int base = blockIdx.x * 512;
  sc[i] = out[MM * 4 + base + i];
  for (int k = 0; k < 4; ++k) bx[i * 4 + k] = out[(base + i) * 4 + k];
  __syncthreads();
  float si = sc[i];
  int r = 0;
  for (int j = 0; j < 512; ++j) {
    float sj = sc[j];
    r += (sj > si) || (sj == si && j < i);      // stable descending rank
  }
  ordIdx[r] = i;
  __syncthreads();
  int src = ordIdx[i];
  float X1 = bx[src * 4 + 0], Y1 = bx[src * 4 + 1];
  float X2 = bx[src * 4 + 2], Y2 = bx[src * 4 + 3];
  sb[i * 4 + 0] = X1; sb[i * 4 + 1] = Y1; sb[i * 4 + 2] = X2; sb[i * 4 + 3] = Y2;
  sArea[i] = fmaxf(X2 - X1, 0.f) * fmaxf(Y2 - Y1, 0.f);
  keepS[i] = 1;
  __syncthreads();
  for (int ii = 0; ii < 511; ++ii) {
    if (i > ii && keepS[ii]) {
      float xx1 = fmaxf(sb[ii * 4 + 0], sb[i * 4 + 0]);
      float yy1 = fmaxf(sb[ii * 4 + 1], sb[i * 4 + 1]);
      float xx2 = fminf(sb[ii * 4 + 2], sb[i * 4 + 2]);
      float yy2 = fminf(sb[ii * 4 + 3], sb[i * 4 + 3]);
      float inter = fmaxf(xx2 - xx1, 0.f) * fmaxf(yy2 - yy1, 0.f);
      float iou = inter / fmaxf(sArea[ii] + sArea[i] - inter, 1e-9f);
      if (iou > 0.5f) keepS[i] = 0;
    }
    __syncthreads();
  }
  out[MM * 4 + MM + base + ordIdx[i]] = keepS[i] ? 1.f : 0.f;
}

// ---------------------------------------------------------------------------
extern "C" void kernel_launch(void* const* d_in, const int* in_sizes, int n_in,
                              void* d_out, int out_size, void* d_ws, size_t ws_size,
                              hipStream_t stream) {
  const float* bfq  = (const float*)d_in[0];
  const float* bfs  = (const float*)d_in[1];
  const float* prop = (const float*)d_in[2];
  const int*   imsz = (const int*)d_in[3];
  const float* W1   = (const float*)d_in[4];
  const float* W2   = (const float*)d_in[5];
  const float* W3   = (const float*)d_in[6];
  const float* Wcor = (const float*)d_in[7];
  const float* fc1w = (const float*)d_in[8];
  const float* fc1b = (const float*)d_in[9];
  const float* fc2w = (const float*)d_in[10];
  const float* fc2b = (const float*)d_in[11];
  const float* sfcw = (const float*)d_in[12];
  const float* sfcb = (const float*)d_in[13];
  const float* scrw = (const float*)d_in[14];
  const float* scrb = (const float*)d_in[15];
  const float* sprw = (const float*)d_in[16];
  const float* sprb = (const float*)d_in[17];
  const float* dprw = (const float*)d_in[18];
  const float* dprb = (const float*)d_in[19];
  float* out = (float*)d_out;

  // Workspace layout (bytes) ~7.6 MB
  char* W = (char*)d_ws;
  size_t off = 0;
  unsigned short* wcomb  = (unsigned short*)(W + off); off += 320 * 256 * 2;   // 163840
  unsigned short* fc1b16 = (unsigned short*)(W + off); off += 256 * 512 * 2;   // 262144
  unsigned short* fc2b16 = (unsigned short*)(W + off); off += 256 * 256 * 2;   // 131072
  float* smean   = (float*)(W + off); off += BB * 256 * 4;                     // 8192
  float* ysw     = (float*)(W + off); off += (size_t)BB * 256 * 64 * 4;        // 524288
  float* w1s     = (float*)(W + off); off += (size_t)BB * 64 * HW * 4;         // 100352
  float* qmean   = (float*)(W + off); off += (size_t)MM * 256 * 4;             // 4194304
  unsigned short* h1b16 = (unsigned short*)(W + off); off += (size_t)MM * 256 * 2; // 2097152
  float* scorefc = (float*)(W + off); off += MM * 4;                           // 16384
  float* spart   = (float*)(W + off); off += MM * 4;                           // 16384
  float* deltas  = (float*)(W + off); off += MM * 4 * 4;                       // 65536
  (void)ws_size; (void)in_sizes; (void)n_in; (void)out_size;

  k_prep_weights<<<512, 256, 0, stream>>>(Wcor, W1, fc1w, fc2w, wcomb, fc1b16, fc2b16);
  k_prep_smean<<<BB, 256, 0, stream>>>(bfs, smean);
  k_prep_ys<<<BB * 256, 64, 0, stream>>>(Wcor, bfs, ysw);
  k_prep_w1s<<<BB * 64, 64, 0, stream>>>(W1, bfs, w1s);

  k_main<<<MM, 256, 0, stream>>>(bfq, wcomb, ysw, w1s, W2, W3,
                                 scrw, scrb, sprw, sprb, dprw, dprb,
                                 qmean, spart, deltas);

  k_init_scorefc<<<16, 256, 0, stream>>>(sfcb, scorefc);
  k_fc1<<<256, 256, 0, stream>>>(qmean, smean, fc1b16, fc1b, h1b16);
  k_fc2<<<256, 256, 0, stream>>>(h1b16, fc2b16, fc2b, sfcw, scorefc);

  k_decode<<<16, 256, 0, stream>>>(spart, scorefc, deltas, prop, imsz, out);
  k_nms<<<BB, 512, 0, stream>>>(out);
}
